// FractionalOrderMaskGenerator_53584011985294
// MI455X (gfx1250) — matrix operand order repaired and checked by reading, not compiled, not run on hardware
//
#include <hip/hip_runtime.h>
#include <hip/hip_bf16.h>
#include <math.h>

#define DEV __device__ __forceinline__

typedef __attribute__((ext_vector_type(16))) _Float16 v16h;
typedef __attribute__((ext_vector_type(8)))  _Float16 v8h;
typedef __attribute__((ext_vector_type(8)))  float    v8f;
typedef __attribute__((ext_vector_type(4)))  unsigned int u32x4;
typedef __attribute__((ext_vector_type(8)))  int      i32x8;
typedef __attribute__((ext_vector_type(4)))  int      i32x4;

#ifndef USE_TDM
#define USE_TDM 1
#endif
#if __has_include(<hip/amd_detail/amd_gfx1250_TDM.h>)
#define TDM_SIX_ARGS 1
#else
#define TDM_SIX_ARGS 0
#endif

constexpr int Bn   = 8;
constexpr int Hh   = 160;
constexpr int Ww   = 160;
constexpr int HWc  = Hh * Ww;          // 25600
constexpr int NPIX = Bn * HWc;         // 204800
constexpr int KT   = 9 * 64;           // 576
constexpr int KHALF = KT / 2;          // 288

DEV v16h concat16(v8h lo, v8h hi) {
  return __builtin_shufflevector(lo, hi, 0,1,2,3,4,5,6,7,8,9,10,11,12,13,14,15);
}

// Stage nElem f16 values (contiguous) from global into LDS via the Tensor Data Mover.
// Issued by wave 0 only; caller is responsible for a workgroup barrier before use.
DEV void stage_lds_1d(_Float16* dst, const _Float16* src, int nElem) {
#if USE_TDM
  if (threadIdx.x < 32) {
    const unsigned long long ga = (unsigned long long)(const void*)src;
    const unsigned ldsa = (unsigned)(unsigned long long)(void*)dst; // low 32b of LDS aperture addr = LDS offset
    u32x4 g0 = { 1u,                                   // count=1, is_restore=0
                 ldsa,                                  // lds_addr
                 (unsigned)(ga & 0xFFFFFFFFull),        // global_addr lo
                 (unsigned)(((ga >> 32) & 0x1FFFFFFull) | (2u << 30)) }; // global hi + type=2
    i32x8 g1 = { 0x10000,                              // data_size=1 (2 bytes), no multicast
                 (int)((nElem & 0xFFFF) << 16),         // tensor_dim0[15:0]
                 (int)((nElem >> 16) & 0xFFFF),         // tensor_dim0[31:16], tensor_dim1 lo = 0
                 (int)((nElem & 0xFFFF) << 16),         // tile_dim0
                 0,                                     // tile_dim1=0, tile_dim2=0
                 nElem,                                 // tensor_dim0_stride lo
                 0, 0 };
    i32x4 z4 = { 0, 0, 0, 0 };
#if TDM_SIX_ARGS
    i32x8 z8 = { 0, 0, 0, 0, 0, 0, 0, 0 };
    __builtin_amdgcn_tensor_load_to_lds(g0, g1, z4, z4, z8, 0);
#else
    __builtin_amdgcn_tensor_load_to_lds(g0, g1, z4, z4, 0);
#endif
    __builtin_amdgcn_s_wait_tensorcnt(0);
  }
#else
  for (int i = (int)threadIdx.x * 8; i < nElem; i += (int)blockDim.x * 8)
    *(v8h*)(dst + i) = *(const v8h*)(src + i);
#endif
}

// ---------------------------------------------------------------------------
// Pack x (NCHW f32) -> xh (NHWC f16) and gray = channel mean (f32).
// Transpose via LDS: coalesced f32 reads along HW, coalesced b128 writes along C.
// ---------------------------------------------------------------------------
__global__ __launch_bounds__(256) void pack_x_kernel(
    const float* __restrict__ x, _Float16* __restrict__ xh, float* __restrict__ gray)
{
  __shared__ _Float16 s[32 * 64];
  __shared__ float gsum[32];
  const int tid = threadIdx.x;
  const int pbase = (int)blockIdx.x * 32;         // 32 pixels; HWc % 32 == 0 so b is uniform
  const int px = tid & 31;
  const int p  = pbase + px;
  const int b  = p / HWc, yx = p % HWc;
  if (tid < 32) gsum[tid] = 0.f;
  __syncthreads();
  float part = 0.f;
  #pragma unroll
  for (int it = 0; it < 8; ++it) {
    const int c = (tid >> 5) + it * 8;
    const float v = x[((size_t)b * 64 + c) * HWc + yx];
    s[px * 64 + c] = (_Float16)v;
    part += v;
  }
  atomicAdd(&gsum[px], part);
  __syncthreads();
  if (tid < 32) gray[pbase + tid] = gsum[tid] * (1.f / 64.f);
  const int prow = tid >> 3, c0 = (tid & 7) * 8;
  *(v8h*)(xh + (size_t)(pbase + prow) * 64 + c0) = *(const v8h*)(s + prow * 64 + c0);
}

// ---------------------------------------------------------------------------
// Pack GEMM weights: w[e][c][3][3] f32 -> wp[khalf][e][288] f16, k = tap*64+c
// ---------------------------------------------------------------------------
__global__ __launch_bounds__(256) void pack_w_kernel(
    const float* __restrict__ w1, const float* __restrict__ w2, const float* __restrict__ w3,
    _Float16* __restrict__ wp1, _Float16* __restrict__ wp2, _Float16* __restrict__ wp3)
{
  const int i = blockIdx.x * blockDim.x + threadIdx.x;
  if (i < 64 * KT) {
    const int e = i / KT, r = i % KT;
    const int tap = r >> 6, c = r & 63;
    const int hs = (r >= KHALF) ? 1 : 0;
    const int dst = hs * (64 * KHALF) + e * KHALF + (r - hs * KHALF);
    wp1[dst] = (_Float16)w1[((size_t)e * 64 + c) * 9 + tap];
    wp2[dst] = (_Float16)w2[((size_t)e * 64 + c) * 9 + tap];
  }
  if (i < 64 * 64) wp3[i] = (_Float16)w3[i];   // [e_out][e_in]
}

// ---------------------------------------------------------------------------
// Implicit-GEMM 3x3 conv (64->64 ch, pad=DIL, dilation=DIL) with WMMA.
// Each wave: 16 pixels x 64 channels. Block = 4 waves = 64 pixels.
// D tiles are swizzled through LDS (reusing the weight buffer) so the block's
// contiguous 8 KB NHWC output region is written with coalesced b128 stores.
// Emits per-channel sum/sumsq (f32) for training-mode batchnorm.
// ---------------------------------------------------------------------------
template <int DIL>
__global__ __launch_bounds__(128) void conv3x3_wmma(
    const _Float16* __restrict__ in,    // [NPIX][64] f16
    const _Float16* __restrict__ wp,    // [2][64][288] f16
    const float*    __restrict__ bias,  // [64]
    _Float16*       __restrict__ out,   // [NPIX][64]
    float*          __restrict__ stats) // [128]: sum[64], sumsq[64]
{
  __shared__ _Float16 lwt[64 * KHALF]; // 36 KB weight half; reused as 8 KB store buffer
  __shared__ float sStats[128];

  const int tid  = threadIdx.x;
  const int wave = tid >> 5, lane = tid & 31;
  const int m = lane & 15, half = lane >> 4;
  const int tileBase = ((int)blockIdx.x * 4 + wave) * 16; // 16-pixel tile, never crosses a row
  const int gp = tileBase + m;
  const int yx = gp % HWc;
  const int y = yx / Ww, x = yx % Ww;
  const _Float16* rowBase = in + (size_t)gp * 64;

  if (tid < 128) sStats[tid] = 0.f;
  __builtin_prefetch(rowBase, 0, 3);
  __builtin_prefetch(wp + 64 * KHALF, 0, 1);

  v8f acc[4] = {};
  for (int stage = 0; stage < 2; ++stage) {
    __syncthreads();                                    // prior reads done before restage
    stage_lds_1d(lwt, wp + stage * (64 * KHALF), 64 * KHALF);
    __syncthreads();
    #pragma unroll
    for (int kl = 0; kl < 9; ++kl) {
      const int kc  = stage * 9 + kl;
      const int tap = kc >> 1;
      const int c0  = (kc & 1) * 32;
      const int dy = tap / 3 - 1, dx = tap % 3 - 1;
      const int yy = y + dy * DIL, xx = x + dx * DIL;
      v16h a = {};
      if (yy >= 0 && yy < Hh && xx >= 0 && xx < Ww) {   // zero-pad halo
        const _Float16* ap = rowBase + (dy * DIL * Ww + dx * DIL) * 64 + c0 + half * 8;
        a = concat16(*(const v8h*)ap, *(const v8h*)(ap + 16));
      }
      const int kloc = kl * 32 + half * 8;
      #pragma unroll
      for (int nt = 0; nt < 4; ++nt) {
        const _Float16* bp = lwt + (nt * 16 + m) * KHALF + kloc;
        v16h bfr = concat16(*(const v8h*)bp, *(const v8h*)(bp + 16));
        acc[nt] = __builtin_amdgcn_wmma_f32_16x16x32_f16(false, a, false, bfr,
                                                         (short)0, acc[nt], false, false);
      }
    }
  }

  // epilogue: + bias, swizzle D through LDS, coalesced b128 stores, BN stats
  __syncthreads();                        // weights dead -> reuse lwt as [64 pix][64 ch] f16
  _Float16* sOut = lwt;
  #pragma unroll
  for (int nt = 0; nt < 4; ++nt) {
    const int ch = nt * 16 + m;
    const float bv = bias[ch];
    float s = 0.f, q = 0.f;
    #pragma unroll
    for (int v = 0; v < 8; ++v) {
      const int pl = wave * 16 + v + 8 * half;          // pixel within block
      const float val = acc[nt][v] + bv;
      sOut[pl * 64 + ch] = (_Float16)val;
      s += val; q += val * val;
    }
    atomicAdd(&sStats[ch], s);
    atomicAdd(&sStats[64 + ch], q);
  }
  __syncthreads();
  {
    _Float16* gout = out + (size_t)blockIdx.x * (64 * 64); // contiguous 8 KB region
    #pragma unroll
    for (int i = tid * 8; i < 64 * 64; i += 128 * 8)
      *(v8h*)(gout + i) = *(const v8h*)(sOut + i);
  }
  if (tid < 128) atomicAdd(&stats[tid], sStats[tid]);
}

// ---------------------------------------------------------------------------
// BN finalize: stats -> per-channel scale/shift
// ---------------------------------------------------------------------------
__global__ void bn_finalize(const float* __restrict__ stats, const float* __restrict__ g,
                            const float* __restrict__ be, float* __restrict__ scsh)
{
  const int e = threadIdx.x;
  if (e < 64) {
    const float inv  = 1.f / (float)NPIX;
    const float mean = stats[e] * inv;
    const float var  = stats[64 + e] * inv - mean * mean;
    const float sc   = g[e] * rsqrtf(var + 1e-5f);
    scsh[e]      = sc;
    scsh[64 + e] = be[e] - mean * sc;
  }
}

// Vectorized BN+ReLU: b128 in / b128 out, 8 f16 per thread
__global__ __launch_bounds__(256) void bn_relu_f16(
    const _Float16* __restrict__ in, const float* __restrict__ scsh, _Float16* __restrict__ out)
{
  const size_t i = ((size_t)blockIdx.x * blockDim.x + threadIdx.x) * 8;
  if (i < (size_t)NPIX * 64) {
    const v8h hv = *(const v8h*)(in + i);
    const int c0 = (int)(i & 63);                 // i is 8-aligned -> channels c0..c0+7
    v8h ov;
    #pragma unroll
    for (int j = 0; j < 8; ++j) {
      const float v = (float)hv[j] * scsh[c0 + j] + scsh[64 + c0 + j];
      ov[j] = (_Float16)fmaxf(v, 0.f);
    }
    *(v8h*)(out + i) = ov;
  }
}

// ---------------------------------------------------------------------------
// Depthwise 3x3 (into LDS) fused with 1x1 conv (WMMA, K=64)
// ---------------------------------------------------------------------------
__global__ __launch_bounds__(128) void dw_pw_wmma(
    const _Float16* __restrict__ in,    // a2 [NPIX][64]
    const float* __restrict__ wd, const float* __restrict__ bd,
    const _Float16* __restrict__ wp3,   // [64][64]
    const float* __restrict__ b3,
    _Float16* __restrict__ out, float* __restrict__ stats)
{
  __shared__ _Float16 lw[64 * 64];
  __shared__ _Float16 dv[64 * 64];     // [pix_local][e]; reused as store buffer
  __shared__ float sStats[128];

  const int tid = threadIdx.x;
  if (tid < 128) sStats[tid] = 0.f;
  stage_lds_1d(lw, wp3, 64 * 64);

  const int blockPix = (int)blockIdx.x * 64;
  for (int idx = tid; idx < 64 * 64; idx += 128) {
    const int pl = idx >> 6, e = idx & 63;
    const int gp = blockPix + pl;
    const int yx = gp % HWc;
    const int y = yx / Ww, x = yx % Ww;
    float a = bd[e];
    #pragma unroll
    for (int t = 0; t < 9; ++t) {
      const int dy = t / 3 - 1, dx = t % 3 - 1;
      const int yy = y + dy, xx = x + dx;
      if (yy >= 0 && yy < Hh && xx >= 0 && xx < Ww)
        a += wd[e * 9 + t] * (float)in[(size_t)(gp + dy * Ww + dx) * 64 + e];
    }
    dv[pl * 64 + e] = (_Float16)a;
  }
  __syncthreads();

  const int wave = tid >> 5, lane = tid & 31;
  const int m = lane & 15, half = lane >> 4;
  v8f acc[4] = {};
  #pragma unroll
  for (int kc = 0; kc < 2; ++kc) {
    const int c0 = kc * 32;
    const _Float16* ap = dv + (wave * 16 + m) * 64 + c0 + half * 8;
    v16h a = concat16(*(const v8h*)ap, *(const v8h*)(ap + 16));
    const int kloc = kc * 32 + half * 8;
    #pragma unroll
    for (int nt = 0; nt < 4; ++nt) {
      const _Float16* bp = lw + (nt * 16 + m) * 64 + kloc;
      v16h bfr = concat16(*(const v8h*)bp, *(const v8h*)(bp + 16));
      acc[nt] = __builtin_amdgcn_wmma_f32_16x16x32_f16(false, a, false, bfr,
                                                       (short)0, acc[nt], false, false);
    }
  }
  __syncthreads();                      // dv reads done -> reuse as store buffer
  _Float16* sOut = dv;
  #pragma unroll
  for (int nt = 0; nt < 4; ++nt) {
    const int ch = nt * 16 + m;
    const float bv = b3[ch];
    float s = 0.f, q = 0.f;
    #pragma unroll
    for (int v = 0; v < 8; ++v) {
      const int pl = wave * 16 + v + 8 * half;
      const float val = acc[nt][v] + bv;
      sOut[pl * 64 + ch] = (_Float16)val;
      s += val; q += val * val;
    }
    atomicAdd(&sStats[ch], s);
    atomicAdd(&sStats[64 + ch], q);
  }
  __syncthreads();
  {
    _Float16* gout = out + (size_t)blockIdx.x * (64 * 64);
    #pragma unroll
    for (int i = tid * 8; i < 64 * 64; i += 128 * 8)
      *(v8h*)(gout + i) = *(const v8h*)(sOut + i);
  }
  if (tid < 128) atomicAdd(&stats[tid], sStats[tid]);
}

// ---------------------------------------------------------------------------
// Head: bn3+relu, 1x1->8 + tanh -> orders, coeff lerp, CIDX gather,
// Sobel direction softmax, attention gate, normalize.
// ---------------------------------------------------------------------------
DEV int cidx_dt(int d, int i, int j) {
  const int dm = d & 3;
  if (dm == 0) { int s = i - 2; return s < 0 ? -s : s; }              // 0, 180 deg
  if (dm == 2) { int s = j - 2; return s < 0 ? -s : s; }              // 90, 270 deg
  if (dm == 1) { int s = i + j - 4; if (s < 0) s = -s; return (3 * s + 1) >> 2; } // 45, 225
  { int s = i - j; if (s < 0) s = -s; return (3 * s + 1) >> 2; }      // 135, 315
}

__global__ __launch_bounds__(256) void final_fuse(
    const _Float16* __restrict__ h3, const _Float16* __restrict__ xh,
    const float* __restrict__ gray, const float* __restrict__ scsh3,
    const float* __restrict__ w4, const float* __restrict__ b4,
    const float* __restrict__ wa1, const float* __restrict__ ba1,
    const float* __restrict__ wa2, const float* __restrict__ ba2,
    float* __restrict__ outFused, float* __restrict__ outOrders)
{
  const int p = blockIdx.x * blockDim.x + threadIdx.x;
  if (p >= NPIX) return;
  const int b = p / HWc, yx = p % HWc;
  const int y = yx / Ww, x = yx % Ww;

  float acc4[8], accA[16];
  #pragma unroll
  for (int d = 0; d < 8; ++d) acc4[d] = b4[d];
  #pragma unroll
  for (int j = 0; j < 16; ++j) accA[j] = ba1[j];
  for (int e = 0; e < 64; ++e) {
    const float rv = fmaxf((float)h3[(size_t)p * 64 + e] * scsh3[e] + scsh3[64 + e], 0.f);
    const float xv = (float)xh[(size_t)p * 64 + e];
    #pragma unroll
    for (int d = 0; d < 8; ++d) acc4[d] += w4[d * 64 + e] * rv;
    #pragma unroll
    for (int j = 0; j < 16; ++j) accA[j] += wa1[j * 64 + e] * xv;
  }
  float attz = ba2[0];
  #pragma unroll
  for (int j = 0; j < 16; ++j) attz += wa2[j] * fmaxf(accA[j], 0.f);
  const float att = 1.f / (1.f + expf(-attz));

  // Sobel (cross-correlation, zero pad) on precomputed gray plane
  float g[3][3];
  #pragma unroll
  for (int di = 0; di < 3; ++di)
    #pragma unroll
    for (int dj = 0; dj < 3; ++dj) {
      const int yy = y + di - 1, xx = x + dj - 1;
      g[di][dj] = (yy >= 0 && yy < Hh && xx >= 0 && xx < Ww) ? gray[b * HWc + yy * Ww + xx] : 0.f;
    }
  const float gx = -g[0][0] + g[0][2] - 2.f * g[1][0] + 2.f * g[1][2] - g[2][0] + g[2][2];
  const float gy = -g[0][0] - 2.f * g[0][1] - g[0][2] + g[2][0] + 2.f * g[2][1] + g[2][2];
  const float mag = sqrtf(gx * gx + gy * gy + 1e-8f);
  const float PIc = 3.14159265358979323846f;
  const float gdir = (atan2f(gy, gx) + PIc) * (8.f / (2.f * PIc));
  float sw[8], smax = -1e30f;
  #pragma unroll
  for (int d = 0; d < 8; ++d) {
    float diff = fabsf(gdir - (float)d);
    diff = fminf(diff, 8.f - diff);
    sw[d] = expf(-diff) * (1.f + mag);
    smax = fmaxf(smax, sw[d]);
  }
  float ssum = 0.f;
  #pragma unroll
  for (int d = 0; d < 8; ++d) { sw[d] = expf(sw[d] - smax); ssum += sw[d]; }
  const float sinv = 1.f / ssum;

  float fused[25];
  #pragma unroll
  for (int t = 0; t < 25; ++t) fused[t] = 0.f;
  #pragma unroll
  for (int d = 0; d < 8; ++d) {
    const float od = 2.f * tanhf(acc4[d]);
    outOrders[((size_t)b * 8 + d) * HWc + yx] = od;
    // reproduce table lookup + lerp: evaluate recurrence at the two linspace nodes
    const float norm = fminf(fmaxf((od + 2.f) * 0.25f, 0.f), 1.f);
    const float idxf = norm * 199.f;
    const float lof = floorf(idxf);
    const float wh  = idxf - lof;
    const float vlo = -2.f + lof * (4.f / 199.f);
    const float vhi = -2.f + ceilf(idxf) * (4.f / 199.f);
    float cd[6]; cd[0] = 1.f;
    float clo = 1.f, chi = 1.f;
    #pragma unroll
    for (int mm = 1; mm < 6; ++mm) {
      clo *= -(vlo - (float)(mm - 1)) / (float)mm;
      chi *= -(vhi - (float)(mm - 1)) / (float)mm;
      cd[mm] = (1.f - wh) * clo + wh * chi;
    }
    const float dwd = sw[d] * sinv;
    #pragma unroll
    for (int t = 0; t < 25; ++t)
      fused[t] += cd[cidx_dt(d, t / 5, t % 5)] * dwd;
  }
  float fsum = 0.f;
  #pragma unroll
  for (int t = 0; t < 25; ++t) { fused[t] *= att; fsum += fused[t]; }
  const float fi = 1.f / (fsum + 1e-8f);
  #pragma unroll
  for (int t = 0; t < 25; ++t)
    outFused[((size_t)b * 25 + t) * HWc + yx] = fused[t] * fi;
}

// ---------------------------------------------------------------------------
extern "C" void kernel_launch(void* const* d_in, const int* in_sizes, int n_in,
                              void* d_out, int out_size, void* d_ws, size_t ws_size,
                              hipStream_t stream)
{
  const float* x   = (const float*)d_in[0];
  const float* w1  = (const float*)d_in[1];
  const float* b1  = (const float*)d_in[2];
  const float* g1  = (const float*)d_in[3];
  const float* be1 = (const float*)d_in[4];
  const float* w2  = (const float*)d_in[5];
  const float* b2  = (const float*)d_in[6];
  const float* g2  = (const float*)d_in[7];
  const float* be2 = (const float*)d_in[8];
  const float* wd  = (const float*)d_in[9];
  const float* bd  = (const float*)d_in[10];
  const float* w3  = (const float*)d_in[11];
  const float* b3  = (const float*)d_in[12];
  const float* g3  = (const float*)d_in[13];
  const float* be3 = (const float*)d_in[14];
  const float* w4  = (const float*)d_in[15];
  const float* b4  = (const float*)d_in[16];
  const float* wa1 = (const float*)d_in[17];
  const float* ba1 = (const float*)d_in[18];
  const float* wa2 = (const float*)d_in[19];
  const float* ba2 = (const float*)d_in[20];
  (void)in_sizes; (void)n_in; (void)out_size; (void)ws_size;

  char* ws = (char*)d_ws;
  size_t off = 0;
  auto alloc = [&](size_t bytes) -> char* {
    char* pp = ws + off; off = (off + bytes + 255) & ~(size_t)255; return pp;
  };
  _Float16* xh    = (_Float16*)alloc((size_t)NPIX * 64 * 2);  // 26.2 MB
  float*    gray  = (float*)   alloc((size_t)NPIX * 4);       // 0.8 MB
  _Float16* slotA = (_Float16*)alloc((size_t)NPIX * 64 * 2);  // 26.2 MB
  _Float16* slotB = (_Float16*)alloc((size_t)NPIX * 64 * 2);  // 26.2 MB
  _Float16* wp1   = (_Float16*)alloc((size_t)64 * KT * 2);
  _Float16* wp2   = (_Float16*)alloc((size_t)64 * KT * 2);
  _Float16* wp3   = (_Float16*)alloc((size_t)64 * 64 * 2);
  float*    stats = (float*)   alloc(3 * 128 * 4);
  float*    scsh  = (float*)   alloc(3 * 128 * 4);

  float* outFused  = (float*)d_out;
  float* outOrders = outFused + (size_t)Bn * 25 * HWc;

  hipMemsetAsync(stats, 0, 3 * 128 * 4, stream);
  pack_x_kernel<<<NPIX / 32, 256, 0, stream>>>(x, xh, gray);
  pack_w_kernel<<<(64 * KT + 255) / 256, 256, 0, stream>>>(w1, w2, w3, wp1, wp2, wp3);

  // conv1 (pad 1) -> bn1 stats -> a1
  conv3x3_wmma<1><<<NPIX / 64, 128, 0, stream>>>(xh, wp1, b1, slotA, stats);
  bn_finalize<<<1, 64, 0, stream>>>(stats, g1, be1, scsh);
  bn_relu_f16<<<(NPIX * 64) / (256 * 8), 256, 0, stream>>>(slotA, scsh, slotB);

  // conv2 (pad 2, dilation 2) -> bn2 stats -> a2
  conv3x3_wmma<2><<<NPIX / 64, 128, 0, stream>>>(slotB, wp2, b2, slotA, stats + 128);
  bn_finalize<<<1, 64, 0, stream>>>(stats + 128, g2, be2, scsh + 128);
  bn_relu_f16<<<(NPIX * 64) / (256 * 8), 256, 0, stream>>>(slotA, scsh + 128, slotB);

  // depthwise 3x3 + 1x1 -> bn3 stats
  dw_pw_wmma<<<NPIX / 64, 128, 0, stream>>>(slotB, wd, bd, wp3, b3, slotA, stats + 256);
  bn_finalize<<<1, 64, 0, stream>>>(stats + 256, g3, be3, scsh + 256);

  // fused head -> (fused, orders)
  final_fuse<<<(NPIX + 255) / 256, 256, 0, stream>>>(slotA, xh, gray, scsh + 256,
      w4, b4, wa1, ba1, wa2, ba2, outFused, outOrders);
}
